// GINlayer_12180527252013
// MI455X (gfx1250) — matrix operand order repaired and checked by reading, not compiled, not run on hardware
//
#include <hip/hip_runtime.h>
#include <hip/hip_bf16.h>
#include <math.h>

#define N_NODES 50000
#define N_EDGES 640000
#define D 128
#define BN_EPS 1e-5f

typedef __attribute__((ext_vector_type(16))) _Float16 v16h;
typedef __attribute__((ext_vector_type(8)))  float    v8f;

// ---- order-preserving float<->uint encoding for atomicMax on floats ----
__device__ __forceinline__ unsigned enc_f32(float f) {
  unsigned u = __float_as_uint(f);
  return (u & 0x80000000u) ? ~u : (u | 0x80000000u);
}
__device__ __forceinline__ float dec_f32(unsigned u) {
  return __uint_as_float((u & 0x80000000u) ? (u ^ 0x80000000u) : ~u);
}

__global__ void init_segmax_kernel(unsigned* __restrict__ segmax) {
  int i = blockIdx.x * blockDim.x + threadIdx.x;
  if (i < N_NODES) segmax[i] = 0x007FFFFFu;  // enc(-inf)
}

// one wave (32 lanes) per edge: attn[e] = dot(nh[src]+eh, nh[dst]); segmax[dst] = max
__global__ void attn_kernel(const float* __restrict__ nh, const float* __restrict__ eh,
                            const int* __restrict__ src, const int* __restrict__ dst,
                            float* __restrict__ attn, unsigned* __restrict__ segmax) {
  int e = (blockIdx.x * blockDim.x + threadIdx.x) >> 5;
  int lane = threadIdx.x & 31;
  if (e >= N_EDGES) return;
  int s = src[e], d = dst[e];
  const float* ps = nh + (size_t)s * D;
  const float* pd = nh + (size_t)d * D;
  const float* pe = eh + (size_t)e * D;
  float p = 0.f;
  #pragma unroll
  for (int c = lane; c < D; c += 32) p += (ps[c] + pe[c]) * pd[c];
  #pragma unroll
  for (int off = 16; off; off >>= 1) p += __shfl_xor(p, off);
  if (lane == 0) {
    attn[e] = p;
    atomicMax(&segmax[d], enc_f32(p));
  }
}

// per edge: ex = exp(attn - segmax[dst]); segsum[dst] += ex; attn <- ex
__global__ void softmax_kernel(const int* __restrict__ dst,
                               const unsigned* __restrict__ segmax,
                               float* __restrict__ segsum, float* __restrict__ attn) {
  int e = blockIdx.x * blockDim.x + threadIdx.x;
  if (e >= N_EDGES) return;
  int d = dst[e];
  float ex = __expf(attn[e] - dec_f32(segmax[d]));
  attn[e] = ex;
  atomicAdd(&segsum[d], ex);
}

// one wave per edge: nz[dst] += alpha * nh[src]
__global__ void scatter_kernel(const float* __restrict__ nh,
                               const int* __restrict__ src, const int* __restrict__ dst,
                               const float* __restrict__ ex, const float* __restrict__ segsum,
                               float* __restrict__ nz) {
  int e = (blockIdx.x * blockDim.x + threadIdx.x) >> 5;
  int lane = threadIdx.x & 31;
  if (e >= N_EDGES) return;
  int s = src[e], d = dst[e];
  float alpha = ex[e] / segsum[d];
  const float* ps = nh + (size_t)s * D;
  float* pd = nz + (size_t)d * D;
  #pragma unroll
  for (int c = lane; c < D; c += 32) atomicAdd(&pd[c], ps[c] * alpha);
}

// ---- fused MLP via WMMA: per 16-row tile, H=relu(X@W1+b1); Y=H@W2+b2 ----
// LDS: w1t(16K f16) | w2t(16K f16) | 8 wave-private 16x128 f16 tiles | bn sums
template <bool IS_EDGE>
__global__ __launch_bounds__(256) void mlp_wmma_kernel(
    const float* __restrict__ xin, const float* __restrict__ eps,
    const float* __restrict__ nz,
    const int* __restrict__ src, const int* __restrict__ dst,
    const float* __restrict__ W1, const float* __restrict__ b1,
    const float* __restrict__ W2, const float* __restrict__ b2,
    float* __restrict__ dout, float* __restrict__ gsum, float* __restrict__ gss,
    int nrows) {
  extern __shared__ _Float16 smem[];
  _Float16* w1t   = smem;             // [n][k] = W1[k][n], 16384 f16
  _Float16* w2t   = smem + 16384;     // [n][k] = W2[k][n]
  _Float16* xsall = smem + 32768;     // 8 waves x (16x128)
  float* lsum = (float*)(smem + 49152);  // 128
  float* lss  = lsum + 128;              // 128

  // cooperative: transpose-convert weights fp32 -> f16 into LDS (L2-resident reads)
  for (int i = threadIdx.x; i < D * D; i += 256) {
    int k = i >> 7, n = i & (D - 1);
    w1t[n * D + k] = (_Float16)W1[i];
    w2t[n * D + k] = (_Float16)W2[i];
  }
  if (threadIdx.x < 128) { lsum[threadIdx.x] = 0.f; lss[threadIdx.x] = 0.f; }
  __syncthreads();

  const int wave = threadIdx.x >> 5;
  const int lane = threadIdx.x & 31;
  const int half = lane >> 4;   // 0: lanes 0-15, 1: lanes 16-31
  const int n16  = lane & 15;
  const int row0 = (blockIdx.x * 8 + wave) * 16;
  _Float16* X = xsall + wave * (16 * D);

  if (row0 < nrows) {  // wave-uniform: EXEC all-1s inside (WMMA requirement)
    // stage fused GIN input tile as f16
    for (int r = 0; r < 16; ++r) {
      const int row = row0 + r;
      if (IS_EDGE) {
        const int s = src[row], d = dst[row];
        const float* pe = xin + (size_t)row * D;
        const float* ps = nz + (size_t)s * D;
        const float* pd = nz + (size_t)d * D;
        for (int c = lane; c < D; c += 32)
          X[r * D + c] = (_Float16)((1.f + eps[c]) * pe[c] + ps[c] - pd[c]);
      } else {
        const float* px = xin + (size_t)row * D;
        const float* pn = nz + (size_t)row * D;
        for (int c = lane; c < D; c += 32)
          X[r * D + c] = (_Float16)((1.f + eps[c]) * px[c] + pn[c]);
      }
    }
    if (IS_EDGE)  // speculative prefetch of the next block's streamed rows
      __builtin_prefetch(xin + (size_t)(row0 + 128) * D + lane, 0, 0);

    // ---------------- GEMM1: H = relu(X @ W1 + b1) ----------------
    v16h a0[4];
    #pragma unroll
    for (int kk = 0; kk < 4; ++kk) {   // A 16x32 f16 layout (ISA 7.12.2)
      const _Float16* p = X + n16 * D + kk * 32 + half * 8;
      #pragma unroll
      for (int j = 0; j < 8; ++j) { a0[kk][j] = p[j]; a0[kk][8 + j] = p[16 + j]; }
    }
    #pragma unroll
    for (int nt = 0; nt < 8; ++nt) {
      v8f c;
      #pragma unroll
      for (int j = 0; j < 8; ++j) c[j] = 0.f;
      #pragma unroll
      for (int kk = 0; kk < 4; ++kk) {  // B 32x16: lane=col, K contiguous from W^T
        v16h b;
        const _Float16* p = w1t + (nt * 16 + n16) * D + kk * 32 + half * 8;
        #pragma unroll
        for (int j = 0; j < 8; ++j) { b[j] = p[j]; b[8 + j] = p[16 + j]; }
        c = __builtin_amdgcn_wmma_f32_16x16x32_f16(false, a0[kk], false, b,
                                                   (short)0, c, false, false);
      }
      const float bias = b1[nt * 16 + n16];
      #pragma unroll
      for (int r = 0; r < 8; ++r) {  // C/D: lane<16 -> M=r, lane>=16 -> M=r+8
        float v = c[r] + bias;
        v = v > 0.f ? v : 0.f;
        X[(half * 8 + r) * D + nt * 16 + n16] = (_Float16)v;  // relu back to tile
      }
    }

    // ---------------- GEMM2: Y = H @ W2 + b2 (+ BN partials) ----------------
    v16h a1[4];
    #pragma unroll
    for (int kk = 0; kk < 4; ++kk) {
      const _Float16* p = X + n16 * D + kk * 32 + half * 8;
      #pragma unroll
      for (int j = 0; j < 8; ++j) { a1[kk][j] = p[j]; a1[kk][8 + j] = p[16 + j]; }
    }
    #pragma unroll
    for (int nt = 0; nt < 8; ++nt) {
      v8f c;
      #pragma unroll
      for (int j = 0; j < 8; ++j) c[j] = 0.f;
      #pragma unroll
      for (int kk = 0; kk < 4; ++kk) {
        v16h b;
        const _Float16* p = w2t + (nt * 16 + n16) * D + kk * 32 + half * 8;
        #pragma unroll
        for (int j = 0; j < 8; ++j) { b[j] = p[j]; b[8 + j] = p[16 + j]; }
        c = __builtin_amdgcn_wmma_f32_16x16x32_f16(false, a1[kk], false, b,
                                                   (short)0, c, false, false);
      }
      const float bias = b2[nt * 16 + n16];
      float s = 0.f, q = 0.f;
      #pragma unroll
      for (int r = 0; r < 8; ++r) {
        float v = c[r] + bias;
        dout[(size_t)(row0 + half * 8 + r) * D + nt * 16 + n16] = v;
        s += v; q += v * v;
      }
      s += __shfl_xor(s, 16);  // combine the two half-columns of N
      q += __shfl_xor(q, 16);
      if (half == 0) {
        atomicAdd(&lsum[nt * 16 + n16], s);
        atomicAdd(&lss[nt * 16 + n16], q);
      }
    }
  }
  __syncthreads();
  if (threadIdx.x < 128) {  // one global atomic per column per block
    atomicAdd(&gsum[threadIdx.x], lsum[threadIdx.x]);
    atomicAdd(&gss[threadIdx.x], lss[threadIdx.x]);
  }
}

__global__ void bn_stat_kernel(const float* __restrict__ bnacc,
                               const float* __restrict__ ngamma, const float* __restrict__ nbeta,
                               const float* __restrict__ egamma, const float* __restrict__ ebeta,
                               float* __restrict__ bnsc) {
  int t = threadIdx.x;
  if (t < 128) {
    float m = bnacc[t] / (float)N_NODES;
    float v = bnacc[128 + t] / (float)N_NODES - m * m;
    float sc = ngamma[t] * rsqrtf(v + BN_EPS);
    bnsc[t] = sc;
    bnsc[128 + t] = nbeta[t] - m * sc;
  } else if (t < 256) {
    int c = t - 128;
    float m = bnacc[256 + c] / (float)N_EDGES;
    float v = bnacc[384 + c] / (float)N_EDGES - m * m;
    float sc = egamma[c] * rsqrtf(v + BN_EPS);
    bnsc[256 + c] = sc;
    bnsc[384 + c] = ebeta[c] - m * sc;
  }
}

__global__ void bn_apply_kernel(float* __restrict__ out, const float* __restrict__ bnsc) {
  const size_t total = (size_t)(N_NODES + N_EDGES) * D;
  const size_t node_end = (size_t)N_NODES * D;
  for (size_t i = (size_t)blockIdx.x * blockDim.x + threadIdx.x; i < total;
       i += (size_t)gridDim.x * blockDim.x) {
    int c = (int)(i & (D - 1));
    int off = (i < node_end) ? 0 : 256;
    out[i] = out[i] * bnsc[off + c] + bnsc[off + 128 + c];
  }
}

extern "C" void kernel_launch(void* const* d_in, const int* in_sizes, int n_in,
                              void* d_out, int out_size, void* d_ws, size_t ws_size,
                              hipStream_t stream) {
  const float* nh = (const float*)d_in[0];
  const float* eh = (const float*)d_in[1];
  const int*   ei = (const int*)d_in[2];
  const int* src = ei;
  const int* dst = ei + N_EDGES;
  const float* nf_eps = (const float*)d_in[3];
  const float* ef_eps = (const float*)d_in[4];
  const float* nf_W1 = (const float*)d_in[5];
  const float* nf_b1 = (const float*)d_in[6];
  const float* nf_W2 = (const float*)d_in[7];
  const float* nf_b2 = (const float*)d_in[8];
  const float* ef_W1 = (const float*)d_in[9];
  const float* ef_b1 = (const float*)d_in[10];
  const float* ef_W2 = (const float*)d_in[11];
  const float* ef_b2 = (const float*)d_in[12];
  const float* nf_gamma = (const float*)d_in[13];
  const float* nf_beta  = (const float*)d_in[14];
  const float* ef_gamma = (const float*)d_in[15];
  const float* ef_beta  = (const float*)d_in[16];

  float* out   = (float*)d_out;
  float* out_n = out;
  float* out_e = out + (size_t)N_NODES * D;

  // workspace layout (floats): segmax | segsum | nz | bnacc(512) | attn | bnsc(512)
  float* f = (float*)d_ws;
  unsigned* segmax = (unsigned*)f;                 // N
  float* segsum = f + N_NODES;                     // N
  float* nz     = segsum + N_NODES;                // N*D
  float* bnacc  = nz + (size_t)N_NODES * D;        // 512
  float* attn   = bnacc + 512;                     // E
  float* bnsc   = attn + N_EDGES;                  // 512

  // zero segsum + nz + bnacc (contiguous), init segmax to enc(-inf)
  hipMemsetAsync(segsum, 0,
                 ((size_t)N_NODES + (size_t)N_NODES * D + 512) * sizeof(float), stream);
  init_segmax_kernel<<<(N_NODES + 255) / 256, 256, 0, stream>>>(segmax);

  attn_kernel<<<(N_EDGES * 32) / 256, 256, 0, stream>>>(nh, eh, src, dst, attn, segmax);
  softmax_kernel<<<(N_EDGES + 255) / 256, 256, 0, stream>>>(dst, segmax, segsum, attn);
  scatter_kernel<<<(N_EDGES * 32) / 256, 256, 0, stream>>>(nh, src, dst, attn, segsum, nz);

  const size_t lds_bytes = (size_t)49152 * sizeof(_Float16) + 256 * sizeof(float);
  mlp_wmma_kernel<false><<<(N_NODES / 16 + 7) / 8, 256, lds_bytes, stream>>>(
      nh, nf_eps, nz, nullptr, nullptr, nf_W1, nf_b1, nf_W2, nf_b2,
      out_n, bnacc + 0, bnacc + 128, N_NODES);
  mlp_wmma_kernel<true><<<(N_EDGES / 16) / 8, 256, lds_bytes, stream>>>(
      eh, ef_eps, nz, src, dst, ef_W1, ef_b1, ef_W2, ef_b2,
      out_e, bnacc + 256, bnacc + 384, N_EDGES);

  bn_stat_kernel<<<1, 256, 0, stream>>>(bnacc, nf_gamma, nf_beta, ef_gamma, ef_beta, bnsc);
  bn_apply_kernel<<<4096, 256, 0, stream>>>(out, bnsc);
}
